// CondAttLSTM_34282428957054
// MI455X (gfx1250) — matrix operand order repaired and checked by reading, not compiled, not run on hardware
//
#include <hip/hip_runtime.h>

// ---------------------------------------------------------------------------
// CondAttLSTM for MI455X (gfx1250): serial scan, per-step matvecs done with
// v_wmma_f32_16x16x32_bf16 (weights pre-transposed + bf16, resident in L2).
// Inner WMMA loop is branch-free: A rows are replicated (only D row 0 is
// consumed, and D[m] depends only on A[m]), g is stored pre-swizzled so each
// lane's A fragment is one contiguous 32B load.
// ---------------------------------------------------------------------------

typedef __attribute__((ext_vector_type(16))) __bf16 v16bf;
typedef __attribute__((ext_vector_type(8)))  float  v8f;

#define T_N    256
#define IN_N   512
#define H_N    1024
#define CTXL   200
#define CTXD_N 512
#define A_N    512
#define NZ     4096   // 4*H
#define KZ     3584   // H + CTXD + H + H  (order: Uh | Cc | Ph | Hh)

__device__ __forceinline__ unsigned short f2bf(float f) {
  union { float f; unsigned int u; } c; c.f = f;
  unsigned int u = c.u;
  unsigned int r = u + 0x7FFFu + ((u >> 16) & 1u);   // round-to-nearest-even
  return (unsigned short)(r >> 16);
}

// A-fragment swizzle: within each 32-element K tile, swap octets 1 and 2 so
// that the 16 halfs needed by each half-wave (K 0-7,16-23 | K 8-15,24-31)
// are contiguous. Octet swap 01<->10 == XOR of bits [4:3] when octet is 1/2.
__device__ __forceinline__ int permA(int k) {
  int o = (k >> 3) & 3;
  return (o == 1 || o == 2) ? (k ^ 24) : k;
}

__device__ __forceinline__ float sigmoidf(float x) {
  return 1.0f / (1.0f + __expf(-x));
}

// WMMA matvec: out[nbase..nbase+15] = g . Wt-rows, with Wt = [N][K] bf16
// (original weight transposed) and gA the swizzled bf16 vector (see permA).
// One wave -> 16 consecutive outputs. All A rows carry the same data (rows
// 1..15 of D are ignored), so there is no divergence in the loop.
__device__ __forceinline__ float wmma_matvec16(const __bf16* __restrict__ gA,
                                               const __bf16* __restrict__ Wt,
                                               int K, int nbase) {
  const int lane = threadIdx.x & 31;
  const int half = lane >> 4;            // 0: K 0-7,16-23 of tile, 1: K 8-15,24-31
  const int l15  = lane & 15;
  const __bf16* arow = gA + half * 16;
  const __bf16* brow = Wt + (size_t)(nbase + l15) * (size_t)K + half * 8;

  v8f acc = {};
  for (int k0 = 0; k0 < K; k0 += 32) {
    v16bf a = *(const v16bf*)(arow + k0);            // 32B uniform per half
    typedef __attribute__((ext_vector_type(8))) __bf16 v8bf;
    v8bf blo = *(const v8bf*)(brow + k0);
    v8bf bhi = *(const v8bf*)(brow + k0 + 16);
    v16bf b = __builtin_shufflevector(blo, bhi, 0, 1, 2, 3, 4, 5, 6, 7, 8, 9, 10, 11, 12, 13, 14, 15);
    __builtin_prefetch(brow + k0 + 128, 0, 1);       // global_prefetch_b8
    acc = __builtin_amdgcn_wmma_f32_16x16x32_bf16(false, a, false, b,
                                                  (short)0, acc, false, false);
  }
  return acc[0];   // VGPR0, lanes 0..15 == D row M=0, N=lane
}

// ---------------- one-time precompute ----------------

// Transpose + bf16-convert all per-step weights. Wz_t stacks Uh|Cc|Ph|Hh to
// match g = [h, ctx_vec, par_h, h_ctx].
__global__ void prep_weights(const float* __restrict__ Wh,
                             const float* __restrict__ Whh,
                             const float* __restrict__ Whist,
                             const float* __restrict__ Uh,
                             const float* __restrict__ Cc,
                             const float* __restrict__ Ph,
                             const float* __restrict__ Hh,
                             unsigned short* __restrict__ Wh_t,
                             unsigned short* __restrict__ Whh_t,
                             unsigned short* __restrict__ Whist_t,
                             unsigned short* __restrict__ Wz_t) {
  const long seg = (long)A_N * H_N;                 // 524288
  const long total = 3 * seg + (long)NZ * KZ;
  for (long idx = (long)blockIdx.x * blockDim.x + threadIdx.x; idx < total;
       idx += (long)gridDim.x * blockDim.x) {
    if (idx < seg) {
      int a = (int)(idx >> 10), k = (int)(idx & 1023);
      Wh_t[idx] = f2bf(Wh[(size_t)k * A_N + a]);
    } else if (idx < 2 * seg) {
      long j = idx - seg;
      int a = (int)(j >> 10), k = (int)(j & 1023);
      Whh_t[j] = f2bf(Whh[(size_t)k * A_N + a]);
    } else if (idx < 3 * seg) {
      long j = idx - 2 * seg;
      int a = (int)(j >> 10), k = (int)(j & 1023);
      Whist_t[j] = f2bf(Whist[(size_t)k * A_N + a]);
    } else {
      long j = idx - 3 * seg;
      int n = (int)(j / KZ), k = (int)(j % KZ);
      float val;
      if (k < H_N)                    val = Uh[(size_t)k * NZ + n];
      else if (k < H_N + CTXD_N)      val = Cc[(size_t)(k - H_N) * NZ + n];
      else if (k < 2 * H_N + CTXD_N)  val = Ph[(size_t)(k - H_N - CTXD_N) * NZ + n];
      else                            val = Hh[(size_t)(k - 2 * H_N - CTXD_N) * NZ + n];
      Wz_t[j] = f2bf(val);
    }
  }
}

// ctx_trans = context @ Wctx + bctx   (one-time, exact fp32)
__global__ void prep_ctx(const float* __restrict__ context,
                         const float* __restrict__ Wctx,
                         const float* __restrict__ bctx,
                         float* __restrict__ ctx_trans) {
  int idx = blockIdx.x * blockDim.x + threadIdx.x;
  if (idx >= CTXL * A_N) return;
  int l = idx / A_N, a = idx % A_N;
  float s = bctx[a];
  for (int d = 0; d < CTXD_N; ++d)
    s += context[(size_t)l * CTXD_N + d] * Wctx[(size_t)d * A_N + a];
  ctx_trans[idx] = s;
}

// XWx = X @ Wx  (one-time, exact fp32; coalesced over n)
__global__ void prep_xwx(const float* __restrict__ X,
                         const float* __restrict__ Wx,
                         float* __restrict__ XWx) {
  int t = blockIdx.x >> 4;                       // 16 n-blocks per row
  int n = ((blockIdx.x & 15) << 8) + threadIdx.x;
  __shared__ float xs[IN_N];
  for (int d = threadIdx.x; d < IN_N; d += blockDim.x)
    xs[d] = X[(size_t)t * IN_N + d];
  __syncthreads();
  float s = 0.0f;
  for (int d = 0; d < IN_N; ++d) s += xs[d] * Wx[(size_t)d * NZ + n];
  XWx[(size_t)t * NZ + n] = s;
}

__global__ void init_state(const float* __restrict__ h0,
                           unsigned short* __restrict__ g_bf) {
  for (int i = threadIdx.x; i < H_N; i += blockDim.x) {
    g_bf[permA(i)] = f2bf(h0[i]);                 // g.h  <- h0
    g_bf[permA(H_N + CTXD_N + i)] = 0;            // g.par <- 0 for t = 0
  }
}

// ---------------- per-step kernels ----------------

// s1 = h@Wh, s2 = h@Whh, histW[t-1] = h@Whist   (h == g_bf[0:H) bf16)
__global__ void __launch_bounds__(128)
step_matvecA(const unsigned short* __restrict__ g_,
             const unsigned short* __restrict__ Wh_t_,
             const unsigned short* __restrict__ Whh_t_,
             const unsigned short* __restrict__ Whist_t_,
             float* __restrict__ s1, float* __restrict__ s2,
             float* __restrict__ histW, int t) {
  int wave  = (blockIdx.x * blockDim.x + threadIdx.x) >> 5;  // 0..95
  int lane  = threadIdx.x & 31;
  int sel   = wave >> 5;
  int nbase = (wave & 31) * 16;
  const __bf16* Wt; float* out;
  if (sel == 0)      { Wt = (const __bf16*)Wh_t_;  out = s1; }
  else if (sel == 1) { Wt = (const __bf16*)Whh_t_; out = s2; }
  else {
    if (t == 0) return;                                      // no history yet
    Wt = (const __bf16*)Whist_t_; out = histW + (size_t)(t - 1) * A_N;
  }
  float r = wmma_matvec16((const __bf16*)g_, Wt, H_N, nbase);
  if (lane < 16) out[nbase + lane] = r;
}

// block 0: context attention -> out_ctx[t], g.ctx
// block 1: history attention -> g.h_ctx
__global__ void __launch_bounds__(256)
step_attention(const float* __restrict__ s1, const float* __restrict__ s2,
               const float* __restrict__ ctx_trans,
               const float* __restrict__ context,
               const float* __restrict__ vvec, const float* __restrict__ bvp,
               const float* __restrict__ bhist, const float* __restrict__ v2,
               const float* __restrict__ bv2p,
               const float* __restrict__ histW, const float* __restrict__ hist,
               float* __restrict__ out_ctx_t, unsigned short* __restrict__ g_bf,
               int t) {
  __shared__ float sv[A_N];
  __shared__ float red[256];
  __shared__ float wgt[256];
  const int tid = threadIdx.x;

  if (blockIdx.x == 0) {
    for (int d = tid; d < A_N; d += 256) sv[d] = s1[d];
    __syncthreads();
    float e = -1e30f;
    if (tid < CTXL) {
      float acc = 0.0f;
      const float* row = ctx_trans + (size_t)tid * A_N;
      for (int d = 0; d < A_N; ++d) acc += tanhf(row[d] + sv[d]) * vvec[d];
      e = acc + bvp[0];
    }
    red[tid] = e; __syncthreads();
    for (int s = 128; s > 0; s >>= 1) {
      if (tid < s) red[tid] = fmaxf(red[tid], red[tid + s]);
      __syncthreads();
    }
    float m = red[0]; __syncthreads();
    float ex = (tid < CTXL) ? __expf(e - m) : 0.0f;
    red[tid] = ex; __syncthreads();
    for (int s = 128; s > 0; s >>= 1) {
      if (tid < s) red[tid] += red[tid + s];
      __syncthreads();
    }
    float inv = 1.0f / red[0];
    __syncthreads();
    wgt[tid] = ex * inv;
    __syncthreads();
    for (int d = tid; d < CTXD_N; d += 256) {
      float cv = 0.0f;
      for (int l = 0; l < CTXL; ++l) cv += wgt[l] * context[(size_t)l * CTXD_N + d];
      out_ctx_t[d] = cv;
      g_bf[permA(H_N + d)] = f2bf(cv);             // g.ctx
    }
  } else {
    if (t == 0) {
      for (int d = tid; d < H_N; d += 256)
        g_bf[permA(H_N + CTXD_N + H_N + d)] = 0;
      return;
    }
    for (int d = tid; d < A_N; d += 256) sv[d] = s2[d] + bhist[d];
    __syncthreads();
    float e = -1e30f;
    if (tid < t) {
      float acc = 0.0f;
      const float* row = histW + (size_t)tid * A_N;
      for (int d = 0; d < A_N; ++d) acc += tanhf(row[d] + sv[d]) * v2[d];
      e = acc + bv2p[0];
    }
    red[tid] = e; __syncthreads();
    for (int s = 128; s > 0; s >>= 1) {
      if (tid < s) red[tid] = fmaxf(red[tid], red[tid + s]);
      __syncthreads();
    }
    float m = red[0]; __syncthreads();
    float ex = (tid < t) ? __expf(e - m) : 0.0f;
    red[tid] = ex; __syncthreads();
    for (int s = 128; s > 0; s >>= 1) {
      if (tid < s) red[tid] += red[tid + s];
      __syncthreads();
    }
    float inv = 1.0f / red[0];
    __syncthreads();
    wgt[tid] = ex * inv;
    __syncthreads();
    for (int d = tid; d < H_N; d += 256) {
      float hv = 0.0f;
      for (int j = 0; j < t; ++j) hv += wgt[j] * hist[(size_t)j * H_N + d];
      g_bf[permA(H_N + CTXD_N + H_N + d)] = f2bf(hv);   // g.h_ctx
    }
  }
}

// zsum = g @ Wz   (K=3584, N=4096) — the big per-step WMMA matvec
__global__ void __launch_bounds__(128)
step_gate(const unsigned short* __restrict__ g_,
          const unsigned short* __restrict__ Wz_t_,
          float* __restrict__ zsum) {
  int wave  = (blockIdx.x * blockDim.x + threadIdx.x) >> 5;  // 0..255
  int lane  = threadIdx.x & 31;
  int nbase = wave * 16;
  float r = wmma_matvec16((const __bf16*)g_, (const __bf16*)Wz_t_, KZ, nbase);
  if (lane < 16) zsum[nbase + lane] = r;
}

// gates -> h_new; write hist[t], out_h[t], g.h; prepare g.par for step t+1
__global__ void __launch_bounds__(256)
step_epilogue(const float* __restrict__ zsum, const float* __restrict__ XWx,
              const float* __restrict__ b4, const float* __restrict__ c0,
              float* __restrict__ hist, float* __restrict__ out_h_t,
              unsigned short* __restrict__ g_bf, const int* __restrict__ parent,
              int t) {
  const float* xw = XWx + (size_t)t * NZ;
  int pnext = (t + 1 < T_N) ? parent[t + 1] : 0;
  for (int i = threadIdx.x; i < H_N; i += 256) {
    float zi = zsum[i]           + xw[i]           + b4[i];
    float zf = zsum[H_N + i]     + xw[H_N + i]     + b4[H_N + i];
    float zc = zsum[2 * H_N + i] + xw[2 * H_N + i] + b4[2 * H_N + i];
    float zo = zsum[3 * H_N + i] + xw[3 * H_N + i] + b4[3 * H_N + i];
    float cn = sigmoidf(zf) * c0[i] + sigmoidf(zi) * tanhf(zc);
    float hn = sigmoidf(zo) * tanhf(cn);
    out_h_t[i] = hn;
    hist[(size_t)t * H_N + i] = hn;
    g_bf[permA(i)] = f2bf(hn);                   // g.h for next step
    if (t + 1 < T_N) {
      float ph = (pnext == t) ? hn : hist[(size_t)pnext * H_N + i];
      g_bf[permA(H_N + CTXD_N + i)] = f2bf(ph);  // g.par for next step
    }
  }
}

// ---------------------------------------------------------------------------

extern "C" void kernel_launch(void* const* d_in, const int* in_sizes, int n_in,
                              void* d_out, int out_size, void* d_ws, size_t ws_size,
                              hipStream_t stream) {
  (void)in_sizes; (void)n_in; (void)out_size; (void)ws_size;

  const float* X       = (const float*)d_in[0];
  const float* context = (const float*)d_in[1];
  const float* h0      = (const float*)d_in[2];
  const float* c0      = (const float*)d_in[3];
  const int*   parent  = (const int*)  d_in[4];
  const float* Wx      = (const float*)d_in[5];
  const float* Uh      = (const float*)d_in[6];
  const float* Cc      = (const float*)d_in[7];
  const float* Ph      = (const float*)d_in[8];
  const float* Hh      = (const float*)d_in[9];
  const float* b4      = (const float*)d_in[10];
  const float* Wctx    = (const float*)d_in[11];
  const float* bctx    = (const float*)d_in[12];
  const float* Wh      = (const float*)d_in[13];
  const float* v       = (const float*)d_in[14];
  const float* bv      = (const float*)d_in[15];
  const float* Whist   = (const float*)d_in[16];
  const float* bhist   = (const float*)d_in[17];
  const float* Whh     = (const float*)d_in[18];
  const float* v2      = (const float*)d_in[19];
  const float* bv2     = (const float*)d_in[20];

  float* out_h   = (float*)d_out;                  // [T][H]
  float* out_ctx = out_h + (size_t)T_N * H_N;      // [T][CTXD]

  // workspace partition (256B aligned slices)
  char* p = (char*)d_ws;
  auto alloc = [&](size_t bytes) {
    char* r = p; p += (bytes + 255) & ~(size_t)255; return r;
  };
  float* ctx_trans = (float*)alloc((size_t)CTXL * A_N * 4);
  float* XWx       = (float*)alloc((size_t)T_N * NZ * 4);
  float* hist      = (float*)alloc((size_t)T_N * H_N * 4);
  float* histW     = (float*)alloc((size_t)T_N * A_N * 4);
  float* s1        = (float*)alloc((size_t)A_N * 4);
  float* s2        = (float*)alloc((size_t)A_N * 4);
  float* zsum      = (float*)alloc((size_t)NZ * 4);
  unsigned short* g_bf    = (unsigned short*)alloc((size_t)KZ * 2);
  unsigned short* Wh_t    = (unsigned short*)alloc((size_t)A_N * H_N * 2);
  unsigned short* Whh_t   = (unsigned short*)alloc((size_t)A_N * H_N * 2);
  unsigned short* Whist_t = (unsigned short*)alloc((size_t)A_N * H_N * 2);
  unsigned short* Wz_t    = (unsigned short*)alloc((size_t)NZ * KZ * 2);

  // one-time precompute (weights land in L2 and stay there: ~34 MB bf16)
  prep_weights<<<2048, 256, 0, stream>>>(Wh, Whh, Whist, Uh, Cc, Ph, Hh,
                                         Wh_t, Whh_t, Whist_t, Wz_t);
  prep_ctx<<<(CTXL * A_N + 255) / 256, 256, 0, stream>>>(context, Wctx, bctx,
                                                         ctx_trans);
  prep_xwx<<<T_N * (NZ / 256), 256, 0, stream>>>(X, Wx, XWx);
  init_state<<<1, 256, 0, stream>>>(h0, g_bf);

  // serial scan: stream ordering carries the recurrence
  for (int t = 0; t < T_N; ++t) {
    step_matvecA<<<24, 128, 0, stream>>>(g_bf, Wh_t, Whh_t, Whist_t,
                                         s1, s2, histW, t);
    step_attention<<<2, 256, 0, stream>>>(s1, s2, ctx_trans, context, v, bv,
                                          bhist, v2, bv2, histW, hist,
                                          out_ctx + (size_t)t * CTXD_N, g_bf, t);
    step_gate<<<64, 128, 0, stream>>>(g_bf, Wz_t, zsum);
    step_epilogue<<<1, 256, 0, stream>>>(zsum, XWx, b4, c0, hist,
                                         out_h + (size_t)t * H_N, g_bf,
                                         parent, t);
  }
}
